// SingleStateRNNBase_59261958750750
// MI455X (gfx1250) — matrix operand order repaired and checked by reading, not compiled, not run on hardware
//
#include <hip/hip_runtime.h>

// RNN scan for MI455X (gfx1250, wave32, WMMA).
// Phase 1: xp = x @ W_ih^T + b_ih  (2048 blocks; W_ih B-frags loaded once per block,
//          8 M-tiles per block streamed through a 128x128 LDS stage; bf16 WMMA)
// Phase 2: h_t = tanh(xp_t + h_{t-1} @ W_hh^T + b_hh)  (4 blocks over batch tiles,
//          W_hh resident in VGPRs, h double-buffered in LDS, in-place overwrite of xp,
//          two independent WMMA accumulators to halve the dependent-chain latency)

typedef __attribute__((ext_vector_type(16))) __bf16 v16bf;
typedef __attribute__((ext_vector_type(8)))  float  v8f;

#define T_STEPS 4096
#define BATCH   64
#define IDIM    128
#define HDIM    128
#define LROW    136   // padded LDS row stride in bf16 elems (272B -> conflict-free b128 loads)

__device__ __forceinline__ float fast_tanhf(float x) {
#if __has_builtin(__builtin_amdgcn_tanhf)
  return __builtin_amdgcn_tanhf(x);
#else
  float y;
  asm volatile("v_tanh_f32 %0, %1" : "=v"(y) : "v"(x));
  return y;
#endif
}

// B-matrix fragment (32x16, bf16) for one K-chunk kc from row-major f32 W row n.
// Lane layout: n = lane%16, half = lane/16; element e -> k = 32*kc + 16*half + e.
__device__ __forceinline__ v16bf load_bfrag(const float* __restrict__ Wrow, int kc, int half) {
  const float4* p = (const float4*)(Wrow + 32 * kc + 16 * half);
  float4 f0 = p[0], f1 = p[1], f2 = p[2], f3 = p[3];
  v16bf b;
  b[0]  = (__bf16)f0.x; b[1]  = (__bf16)f0.y; b[2]  = (__bf16)f0.z; b[3]  = (__bf16)f0.w;
  b[4]  = (__bf16)f1.x; b[5]  = (__bf16)f1.y; b[6]  = (__bf16)f1.z; b[7]  = (__bf16)f1.w;
  b[8]  = (__bf16)f2.x; b[9]  = (__bf16)f2.y; b[10] = (__bf16)f2.z; b[11] = (__bf16)f2.w;
  b[12] = (__bf16)f3.x; b[13] = (__bf16)f3.y; b[14] = (__bf16)f3.z; b[15] = (__bf16)f3.w;
  return b;
}

// A-matrix fragment (16x32, bf16) for one K-chunk kc from padded row-major bf16 LDS.
// Lane layout: m = lane%16, half = lane/16.
//   VGPRs 0-3: k = 32*kc + 16*half + [0..7]   (16 contiguous bytes)
//   VGPRs 4-7: k = 32*kc + 16*half + 8 + [0..7]
__device__ __forceinline__ v16bf load_afrag(const __bf16* __restrict__ rowbase, int kc, int half) {
  union { uint4 u[2]; v16bf v; } a;
  a.u[0] = *(const uint4*)(rowbase + 32 * kc + 16 * half);
  a.u[1] = *(const uint4*)(rowbase + 32 * kc + 16 * half + 8);
  return a.v;
}

// ---------------- Phase 1: xp = x @ W_ih^T + b_ih -> out (hs region) ----------------
__global__ void __launch_bounds__(256) rnn_xproj_kernel(
    const float* __restrict__ x, const float* __restrict__ W_ih,
    const float* __restrict__ b_ih, float* __restrict__ xp) {
  __shared__ __align__(16) __bf16 xbuf[128 * LROW];   // 8 M-tiles (128 rows), ~34 KB
  const int tid     = threadIdx.x;
  const size_t row0 = (size_t)blockIdx.x * 128;       // flattened (t,b) row block

  // Stage 128x128 f32 block of x into LDS as bf16 (padded rows). 16 float4 per thread.
  for (int i = tid; i < 4096; i += 256) {
    int r = i >> 5, c4 = (i & 31) << 2;
    const float4 f = *(const float4*)(x + (row0 + r) * IDIM + c4);
    __bf16* dst = &xbuf[r * LROW + c4];
    dst[0] = (__bf16)f.x; dst[1] = (__bf16)f.y; dst[2] = (__bf16)f.z; dst[3] = (__bf16)f.w;
  }

  const int lane = tid & 31;
  const int wv   = tid >> 5;          // wave id 0..7 = N tile
  const int half = lane >> 4;
  const int m    = lane & 15;
  const int n    = (lane & 15) + wv * 16;

  // W_ih B-fragments: loaded ONCE per block, reused for all 8 M-tiles.
  v16bf bfr[4];
#pragma unroll
  for (int kc = 0; kc < 4; ++kc) bfr[kc] = load_bfrag(W_ih + (size_t)n * IDIM, kc, half);
  const float bias = b_ih[n];

  __syncthreads();

#pragma unroll
  for (int mt = 0; mt < 8; ++mt) {
    v16bf afr[4];
#pragma unroll
    for (int kc = 0; kc < 4; ++kc)
      afr[kc] = load_afrag(&xbuf[(mt * 16 + m) * LROW], kc, half);

    v8f c0 = {}, c1 = {};
    c0 = __builtin_amdgcn_wmma_f32_16x16x32_bf16(false, afr[0], false, bfr[0], (short)0, c0, false, false);
    c1 = __builtin_amdgcn_wmma_f32_16x16x32_bf16(false, afr[1], false, bfr[1], (short)0, c1, false, false);
    c0 = __builtin_amdgcn_wmma_f32_16x16x32_bf16(false, afr[2], false, bfr[2], (short)0, c0, false, false);
    c1 = __builtin_amdgcn_wmma_f32_16x16x32_bf16(false, afr[3], false, bfr[3], (short)0, c1, false, false);

    float* outp = xp + (row0 + mt * 16) * HDIM + n;
#pragma unroll
    for (int r = 0; r < 8; ++r)
      outp[(size_t)(r + 8 * half) * HDIM] = c0[r] + c1[r] + bias;
  }
}

// ---------------- Phase 2: sequential scan over T, 4 blocks over batch ----------------
__global__ void __launch_bounds__(256) rnn_scan_kernel(
    const float* __restrict__ h0, const float* __restrict__ W_hh,
    const float* __restrict__ b_hh, float* __restrict__ out) {
  __shared__ __align__(16) __bf16 hbuf[2][16 * LROW];
  const int tid = threadIdx.x;
  const int b0  = blockIdx.x * 16;  // batch tile

  // Stage h0 tile into buffer 0 as bf16.
  for (int i = tid; i < 512; i += 256) {
    int r = i >> 5, c4 = (i & 31) << 2;
    const float4 f = *(const float4*)(h0 + (size_t)(b0 + r) * HDIM + c4);
    __bf16* dst = &hbuf[0][r * LROW + c4];
    dst[0] = (__bf16)f.x; dst[1] = (__bf16)f.y; dst[2] = (__bf16)f.z; dst[3] = (__bf16)f.w;
  }

  const int lane = tid & 31;
  const int wv   = tid >> 5;
  const int half = lane >> 4;
  const int m    = lane & 15;
  const int n    = (lane & 15) + wv * 16;

  // W_hh B-fragments stay resident in VGPRs for the whole scan.
  v16bf bfr[4];
#pragma unroll
  for (int kc = 0; kc < 4; ++kc) bfr[kc] = load_bfrag(W_hh + (size_t)n * HDIM, kc, half);
  const float bias = b_hh[n];

  __syncthreads();

  // Prefetch xp tile for t = 0.
  float xpv[8];
  {
    const float* tile = out + (size_t)b0 * HDIM + n;
#pragma unroll
    for (int r = 0; r < 8; ++r) xpv[r] = tile[(size_t)(r + 8 * half) * HDIM];
  }

  for (int t = 0; t < T_STEPS; ++t) {
    const int p = t & 1;
    v16bf afr[4];
#pragma unroll
    for (int kc = 0; kc < 4; ++kc) afr[kc] = load_afrag(&hbuf[p][m * LROW], kc, half);

    // Two independent accumulator chains (halves the WMMA RAW chain per step).
    v8f c0, c1 = {};
#pragma unroll
    for (int r = 0; r < 8; ++r) c0[r] = xpv[r] + bias;

    c0 = __builtin_amdgcn_wmma_f32_16x16x32_bf16(false, afr[0], false, bfr[0], (short)0, c0, false, false);
    c1 = __builtin_amdgcn_wmma_f32_16x16x32_bf16(false, afr[1], false, bfr[1], (short)0, c1, false, false);
    c0 = __builtin_amdgcn_wmma_f32_16x16x32_bf16(false, afr[2], false, bfr[2], (short)0, c0, false, false);
    c1 = __builtin_amdgcn_wmma_f32_16x16x32_bf16(false, afr[3], false, bfr[3], (short)0, c1, false, false);

    float hv[8];
#pragma unroll
    for (int r = 0; r < 8; ++r) hv[r] = fast_tanhf(c0[r] + c1[r]);

    // Overwrite xp_t tile in d_out with h_t (same wave, same addresses).
    float* tile = out + ((size_t)t * BATCH + b0) * HDIM + n;
#pragma unroll
    for (int r = 0; r < 8; ++r)
      tile[(size_t)(r + 8 * half) * HDIM] = hv[r];

    // Write bf16 h_t into the other LDS buffer for the next step's A-matrix.
    __bf16* wb = hbuf[p ^ 1];
#pragma unroll
    for (int r = 0; r < 8; ++r)
      wb[(r + 8 * half) * LROW + n] = (__bf16)hv[r];

    if (t == T_STEPS - 1) {
      float* fin = out + (size_t)T_STEPS * BATCH * HDIM + (size_t)b0 * HDIM + n;
#pragma unroll
      for (int r = 0; r < 8; ++r)
        fin[(size_t)(r + 8 * half) * HDIM] = hv[r];
    }

    // Prefetch xp for t+1 (t+1 == T reads the tail region: in-bounds, value unused).
    const float* ntile = out + ((size_t)(t + 1) * BATCH + b0) * HDIM + n;
#pragma unroll
    for (int r = 0; r < 8; ++r) xpv[r] = ntile[(size_t)(r + 8 * half) * HDIM];

    __syncthreads();
  }
}

extern "C" void kernel_launch(void* const* d_in, const int* in_sizes, int n_in,
                              void* d_out, int out_size, void* d_ws, size_t ws_size,
                              hipStream_t stream) {
  (void)in_sizes; (void)n_in; (void)out_size; (void)d_ws; (void)ws_size;
  const float* x    = (const float*)d_in[0];
  const float* h0   = (const float*)d_in[1];
  const float* W_ih = (const float*)d_in[2];
  const float* W_hh = (const float*)d_in[3];
  const float* b_ih = (const float*)d_in[4];
  const float* b_hh = (const float*)d_in[5];
  float* out = (float*)d_out;

  // Phase 1: 262144 rows / 128 per block = 2048 blocks, 8 waves each.
  rnn_xproj_kernel<<<(T_STEPS * BATCH) / 128, 256, 0, stream>>>(x, W_ih, b_ih, out);
  // Phase 2: 4 independent batch tiles, 8 waves each (one N-tile per wave).
  rnn_scan_kernel<<<BATCH / 16, 256, 0, stream>>>(h0, W_hh, b_hh, out);
}
